// MultiHeadAttention_23270132809952
// MI455X (gfx1250) — matrix operand order repaired and checked by reading, not compiled, not run on hardware
//
#include <hip/hip_runtime.h>

typedef __attribute__((ext_vector_type(16))) __bf16 v16bf;
typedef __attribute__((ext_vector_type(8)))  __bf16 v8bf;
typedef __attribute__((ext_vector_type(8)))  float  v8f;

#define BATCH 4
#define SEQ   2048
#define DIM   1024
#define NH    16
#define HD    64

__device__ __forceinline__ v16bf frag_cat(v8bf lo, v8bf hi) {
  v16bf r;
#pragma unroll
  for (int i = 0; i < 8; ++i) { r[i] = lo[i]; r[i + 8] = hi[i]; }
  return r;
}

__device__ __forceinline__ v8f wmma_bf16(v16bf a, v16bf b, v8f c) {
  return __builtin_amdgcn_wmma_f32_16x16x32_bf16(false, a, false, b, (short)0, c,
                                                 false, false);
}

// async global->LDS 16B transfer; lds_ptr escapes via ptrtoint so the compiler
// treats LDS as written by the asm (memory clobber), keeping ds_loads live.
__device__ __forceinline__ void async_load_b128(const void* gptr,
                                                const void* lds_ptr) {
  unsigned lds_addr = (unsigned)(size_t)lds_ptr;  // flat addr[31:0] == LDS offset
  asm volatile("global_load_async_to_lds_b128 %0, %1, off"
               :: "v"(lds_addr), "v"((unsigned long long)(size_t)gptr)
               : "memory");
}

// ---------------- f32 -> bf16 converters ----------------
__global__ void f32_to_bf16(const float* __restrict__ src,
                            __bf16* __restrict__ dst, int n) {
  for (int i = blockIdx.x * blockDim.x + threadIdx.x; i < n;
       i += gridDim.x * blockDim.x)
    dst[i] = (__bf16)src[i];
}

// dst[n*K + k] = src[k*N + n]   (transpose so B-operand is K-contiguous)
__global__ void f32_to_bf16_t(const float* __restrict__ src,
                              __bf16* __restrict__ dst, int K, int N) {
  for (int i = blockIdx.x * blockDim.x + threadIdx.x; i < K * N;
       i += gridDim.x * blockDim.x) {
    int nn = i / K, k = i - nn * K;
    dst[i] = (__bf16)src[(size_t)k * N + nn];
  }
}

// ---------------- tiled WMMA GEMM: C = A[M,K] * BT[N,K]^T + bias ----------------
// block tile 64(M) x 128(N), k-step 32, 8 waves (4x2 per-wave 16x64 tiles).
// LDS staging uses CDNA5 async global->LDS DMA (ASYNCcnt), double buffered.
#define TILE_ELTS (64 * 32 + 128 * 32)  // bf16 elements per buffer (A + B)

__global__ __launch_bounds__(256) void gemm_proj(
    const __bf16* __restrict__ A, const __bf16* __restrict__ BT,
    const float* __restrict__ bias, float* __restrict__ outF,
    __bf16* __restrict__ outB, __bf16* __restrict__ outT, int M, int N, int K,
    int Sdim) {
  __shared__ __align__(16) __bf16 sAB[2][TILE_ELTS];  // 24 KB
  const int tid = threadIdx.x;
  const int lane = tid & 31;
  const int wave = tid >> 5;
  const int wm = wave & 3;   // 0..3 -> rows 16*wm
  const int wn = wave >> 2;  // 0..1 -> cols 64*wn
  const int m0 = blockIdx.x * 64;
  const int n0 = blockIdx.y * 128;
  const int krA = (lane >> 4) * 8;   // A frag K base: 0 / 8
  const int krB = (lane >> 4) * 8;   // B frag K base: 0 / 8

  // per-thread staging chunks (16B each): 1x A, 2x B
  const int aIdx = tid * 8;  // 0..2047 -> A[64][32]
  const int aR = aIdx >> 5, aC = aIdx & 31;

  v8f acc[4] = {v8f{}, v8f{}, v8f{}, v8f{}};

  // issue one k-tile's async global->LDS transfers into buffer `buf`
  auto issue_tile = [&](int k0, int buf) {
    async_load_b128(A + (size_t)(m0 + aR) * K + k0 + aC, &sAB[buf][aIdx]);
#pragma unroll
    for (int t = 0; t < 2; ++t) {
      int idx = (tid + t * 256) * 8;  // 0..4095 -> B[128][32]
      int r = idx >> 5, c = idx & 31;
      async_load_b128(BT + (size_t)(n0 + r) * K + k0 + c,
                      &sAB[buf][64 * 32 + idx]);
    }
  };

  issue_tile(0, 0);
  int it = 0;
  for (int k0 = 0; k0 < K; k0 += 32, ++it) {
    const int buf = it & 1;
    if (k0 + 32 < K) {
      issue_tile(k0 + 32, buf ^ 1);                    // prefetch next tile
      asm volatile("s_wait_asynccnt 3" ::: "memory");  // this tile's 3 done
    } else {
      asm volatile("s_wait_asynccnt 0" ::: "memory");
    }
    __syncthreads();

    const __bf16* sA = &sAB[buf][0];
    const __bf16* sB = &sAB[buf][64 * 32];
    const int arow = wm * 16 + (lane & 15);
    v16bf af = frag_cat(*(const v8bf*)(sA + arow * 32 + krA),
                        *(const v8bf*)(sA + arow * 32 + krA + 16));
#pragma unroll
    for (int t = 0; t < 4; ++t) {
      int bcol = wn * 64 + t * 16 + (lane & 15);
      v16bf bf_ = frag_cat(*(const v8bf*)(sB + bcol * 32 + krB),
                           *(const v8bf*)(sB + bcol * 32 + krB + 16));
      acc[t] = wmma_bf16(af, bf_, acc[t]);
    }
    __syncthreads();  // protect buf against next iteration's async writes
  }

  const int rbase = m0 + wm * 16 + (lane >> 4) * 8;
#pragma unroll
  for (int t = 0; t < 4; ++t) {
    int col = n0 + wn * 64 + t * 16 + (lane & 15);
    float bv = bias ? bias[col] : 0.0f;
#pragma unroll
    for (int i = 0; i < 8; ++i) {
      int row = rbase + i;
      float v = acc[t][i] + bv;
      if (outF) outF[(size_t)row * N + col] = v;
      if (outB) outB[(size_t)row * N + col] = (__bf16)v;
      if (outT) {  // row = b*S + s ; col = h*HD + d -> VT[b][h][d][s]
        int bb = row / Sdim, s = row - bb * Sdim;
        int h = col >> 6, d = col & 63;
        outT[((((size_t)bb * NH) + h) * HD + d) * (size_t)Sdim + s] = (__bf16)v;
      }
    }
  }
}

// ---------------- fused scores + softmax + context ----------------
// grid: (SEQ/16, NH, BATCH). 256 threads. dyn LDS: 16*SEQ f32 scores + 2*16*64 ctx
__global__ __launch_bounds__(256) void attn_kernel(
    const __bf16* __restrict__ Qb,  // [B*S, D]
    const __bf16* __restrict__ Kb,  // [B*S, D]
    const __bf16* __restrict__ VT,  // [B, H, HD, S]
    const float* __restrict__ mask, // [B, 1, S, S]
    float* __restrict__ align_out,  // [B, H, S, S]
    __bf16* __restrict__ ctx_out)   // [B*S, D]
{
  extern __shared__ char smem_raw[];
  float* P = (float*)smem_raw;    // [16][SEQ]
  float* CTX = P + 16 * SEQ;      // [2][16][64]

  const int tid = threadIdx.x;
  const int lane = tid & 31;
  const int wave = tid >> 5;
  const int b = blockIdx.z, h = blockIdx.y, q0 = blockIdx.x * 16;
  const int krA = (lane >> 4) * 8;
  const int krB = (lane >> 4) * 8;

  // ---- phase 1: scores = Q K^T * scale, mask, into LDS ----
  const __bf16* qrow =
      Qb + (size_t)(b * SEQ + q0 + (lane & 15)) * DIM + h * HD;
  v16bf aq0 = frag_cat(*(const v8bf*)(qrow + krA),
                       *(const v8bf*)(qrow + krA + 16));
  v16bf aq1 = frag_cat(*(const v8bf*)(qrow + 32 + krA),
                       *(const v8bf*)(qrow + 32 + krA + 16));
#pragma unroll 4
  for (int nt = 0; nt < 16; ++nt) {
    int col = wave * 256 + nt * 16 + (lane & 15);
    const __bf16* krow = Kb + (size_t)(b * SEQ + col) * DIM + h * HD;
    v16bf bk0 = frag_cat(*(const v8bf*)(krow + krB),
                         *(const v8bf*)(krow + krB + 16));
    v16bf bk1 = frag_cat(*(const v8bf*)(krow + 32 + krB),
                         *(const v8bf*)(krow + 32 + krB + 16));
    v8f acc = {};
    acc = wmma_bf16(aq0, bk0, acc);
    acc = wmma_bf16(aq1, bk1, acc);
#pragma unroll
    for (int i = 0; i < 8; ++i) {
      int ql = i + (lane >> 4) * 8;
      float s = acc[i] * 0.125f;  // 1/sqrt(64)
      float mv = mask[((size_t)b * SEQ + (q0 + ql)) * (size_t)SEQ + col];
      s *= mv;
      if (s == 0.0f) s = -1e9f;
      P[ql * SEQ + col] = s;
    }
  }
  __syncthreads();

  // ---- phase 2: row softmax (2 rows / wave), write alignment ----
#pragma unroll
  for (int rr = 0; rr < 2; ++rr) {
    int r = wave * 2 + rr;
    float* row = P + r * SEQ;
    float m = -3.0e38f;
    for (int c = lane; c < SEQ; c += 32) m = fmaxf(m, row[c]);
#pragma unroll
    for (int off = 16; off; off >>= 1) m = fmaxf(m, __shfl_xor(m, off, 32));
    float sum = 0.0f;
    for (int c = lane; c < SEQ; c += 32) {
      float e = __expf(row[c] - m);
      row[c] = e;
      sum += e;
    }
#pragma unroll
    for (int off = 16; off; off >>= 1) sum += __shfl_xor(sum, off, 32);
    float inv = 1.0f / sum;
    float* aout =
        align_out + (((size_t)(b * NH + h)) * SEQ + (q0 + r)) * (size_t)SEQ;
    for (int c = lane; c < SEQ; c += 32) {
      float p = row[c] * inv;
      row[c] = p;
      aout[c] = p;
    }
  }
  __syncthreads();

  // ---- phase 3: context = P @ V  (waves: 4 n-tiles x 2 k-halves) ----
  {
    const int nt4 = wave & 3;
    const int kh = wave >> 2;
    const int arow = lane & 15;
    const int dcol = nt4 * 16 + (lane & 15);
    const __bf16* vbase =
        VT + (((size_t)(b * NH + h)) * HD + dcol) * (size_t)SEQ;
    v8f acc = {};
    for (int kk = kh * 1024; kk < kh * 1024 + 1024; kk += 32) {
      const float* ap = P + arow * SEQ + kk + krA;
      v16bf af;
#pragma unroll
      for (int j = 0; j < 8; ++j) {
        af[j] = (__bf16)ap[j];
        af[j + 8] = (__bf16)ap[j + 16];
      }
      const __bf16* bp = vbase + kk + krB;
      v16bf bf_ = frag_cat(*(const v8bf*)bp, *(const v8bf*)(bp + 16));
      acc = wmma_bf16(af, bf_, acc);
    }
#pragma unroll
    for (int i = 0; i < 8; ++i) {
      int ql = i + (lane >> 4) * 8;
      CTX[kh * 1024 + ql * 64 + dcol] = acc[i];
    }
  }
  __syncthreads();
  for (int idx = tid; idx < 16 * 64; idx += 256) {
    float v = CTX[idx] + CTX[1024 + idx];
    int ql = idx >> 6, dc = idx & 63;
    ctx_out[(size_t)(b * SEQ + q0 + ql) * DIM + h * HD + dc] = (__bf16)v;
  }
}

// ---------------- host launch ----------------
extern "C" void kernel_launch(void* const* d_in, const int* in_sizes, int n_in,
                              void* d_out, int out_size, void* d_ws,
                              size_t ws_size, hipStream_t stream) {
  const float* query = (const float*)d_in[0];
  const float* value = (const float*)d_in[1];
  const float* mask  = (const float*)d_in[2];
  const float* wq = (const float*)d_in[3];
  const float* bq = (const float*)d_in[4];
  const float* wk = (const float*)d_in[5];
  const float* bk = (const float*)d_in[6];
  const float* wv = (const float*)d_in[7];
  const float* bv = (const float*)d_in[8];
  const float* wo = (const float*)d_in[9];
  const float* bo = (const float*)d_in[10];

  float* heads = (float*)d_out;                             // [B,S,D]
  float* align = heads + (size_t)BATCH * SEQ * DIM;         // [B,H,S,S]

  char* ws = (char*)d_ws;
  size_t off = 0;
  auto alloc = [&](size_t bytes) -> char* {
    char* p = ws + off;
    off += (bytes + 255) & ~(size_t)255;
    return p;
  };
  const size_t MS = (size_t)BATCH * SEQ;  // 8192
  __bf16* qbf = (__bf16*)alloc(MS * DIM * 2);
  __bf16* vbf = (__bf16*)alloc(MS * DIM * 2);
  __bf16* wqT = (__bf16*)alloc((size_t)DIM * DIM * 2);
  __bf16* wkT = (__bf16*)alloc((size_t)DIM * DIM * 2);
  __bf16* wvT = (__bf16*)alloc((size_t)DIM * DIM * 2);
  __bf16* woT = (__bf16*)alloc((size_t)DIM * DIM * 2);
  __bf16* Qb  = (__bf16*)alloc(MS * DIM * 2);
  __bf16* Kb  = (__bf16*)alloc(MS * DIM * 2);
  __bf16* VT  = (__bf16*)alloc(MS * DIM * 2);
  __bf16* ctx = (__bf16*)alloc(MS * DIM * 2);

  f32_to_bf16<<<4096, 256, 0, stream>>>(query, qbf, (int)(MS * DIM));
  f32_to_bf16<<<4096, 256, 0, stream>>>(value, vbf, (int)(MS * DIM));
  f32_to_bf16_t<<<2048, 256, 0, stream>>>(wq, wqT, DIM, DIM);
  f32_to_bf16_t<<<2048, 256, 0, stream>>>(wk, wkT, DIM, DIM);
  f32_to_bf16_t<<<2048, 256, 0, stream>>>(wv, wvT, DIM, DIM);
  f32_to_bf16_t<<<2048, 256, 0, stream>>>(wo, woT, DIM, DIM);

  dim3 ggrid(MS / 64, DIM / 128);
  gemm_proj<<<ggrid, 256, 0, stream>>>(qbf, wqT, bq, nullptr, Qb, nullptr,
                                       (int)MS, DIM, DIM, SEQ);
  gemm_proj<<<ggrid, 256, 0, stream>>>(vbf, wkT, bk, nullptr, Kb, nullptr,
                                       (int)MS, DIM, DIM, SEQ);
  gemm_proj<<<ggrid, 256, 0, stream>>>(vbf, wvT, bv, nullptr, nullptr, VT,
                                       (int)MS, DIM, DIM, SEQ);

  size_t smem = (size_t)16 * SEQ * 4 + (size_t)2 * 16 * 64 * 4;
  (void)hipFuncSetAttribute((const void*)attn_kernel,
                            hipFuncAttributeMaxDynamicSharedMemorySize,
                            (int)smem);
  attn_kernel<<<dim3(SEQ / 16, NH, BATCH), 256, smem, stream>>>(
      Qb, Kb, VT, mask, align, ctx);

  gemm_proj<<<ggrid, 256, 0, stream>>>(ctx, woT, bo, heads, nullptr, nullptr,
                                       (int)MS, DIM, DIM, SEQ);
}
